// VGG_Small_6983616824301
// MI455X (gfx1250) — 8-bit matrix operand order repaired and checked by reading, compiled, not run on hardware
//
#include <hip/hip_runtime.h>
#include <stdint.h>

typedef __attribute__((ext_vector_type(8))) int v8i;

#define EPSV 1e-5f

__device__ __forceinline__ int kbaseB(int i, int hi) {
  return ((i >> 1) << 4) + (hi << 3) + ((i & 1) << 2);
}

__global__ void conv1_kernel(const float* __restrict__ x, const float* __restrict__ w,
                             const float* __restrict__ bias, float* __restrict__ out, int B) {
  int idx = blockIdx.x * blockDim.x + threadIdx.x;
  int total = B * 32 * 32 * 128;
  if (idx >= total) return;
  int co = idx & 127;
  int t = idx >> 7;
  int xo = t & 31; t >>= 5;
  int yo = t & 31; t >>= 5;
  int b = t;
  float acc = bias[co];
  for (int ci = 0; ci < 3; ++ci)
    for (int ky = 0; ky < 3; ++ky) {
      int ys = yo + ky - 1;
      if (ys < 0 || ys >= 32) continue;
      for (int kx = 0; kx < 3; ++kx) {
        int xs = xo + kx - 1;
        if (xs < 0 || xs >= 32) continue;
        acc += x[((b * 3 + ci) * 32 + ys) * 32 + xs] * w[((co * 3 + ci) * 3 + ky) * 3 + kx];
      }
    }
  out[idx] = acc;
}

__global__ void stats_kernel(const float* __restrict__ h, long long n, int C,
                             float* __restrict__ sums) {
  long long idx = (long long)blockIdx.x * blockDim.x + threadIdx.x;
  long long stride = (long long)gridDim.x * blockDim.x;
  int c = (int)(idx % C);
  float s = 0.f, sq = 0.f;
  for (; idx < n; idx += stride) {
    float v = h[idx];
    s += v;
    sq += v * v;
  }
  atomicAdd(&sums[c], s);
  atomicAdd(&sums[C + c], sq);
}

__global__ void bnfin_kernel(const float* __restrict__ sums, const float* __restrict__ g,
                             const float* __restrict__ beta, float invN, int C,
                             float* __restrict__ ab) {
  int c = blockIdx.x * blockDim.x + threadIdx.x;
  if (c >= C) return;
  float mean = sums[c] * invN;
  float var = sums[C + c] * invN - mean * mean;
  float a = g[c] * rsqrtf(var + EPSV);
  ab[c] = a;
  ab[C + c] = beta[c] - mean * a;
}

__global__ void pack_kernel(const float* __restrict__ h, const float* __restrict__ ab,
                            long long n, int C, int8_t* __restrict__ out) {
  long long idx = (long long)blockIdx.x * blockDim.x + threadIdx.x;
  if (idx >= n) return;
  int c = (int)(idx % C);
  float v = h[idx] * ab[c] + ab[C + c];
  out[idx] = (v >= 0.f) ? (int8_t)1 : (int8_t)-1;
}

__global__ void pack_fc1_kernel(const float* __restrict__ h, const float* __restrict__ ab,
                                int B, int8_t* __restrict__ out) {
  int idx = blockIdx.x * blockDim.x + threadIdx.x;
  if (idx >= B * 8192) return;
  int k = idx & 8191;
  int b = idx >> 13;
  int c = k >> 4;
  int y = (k >> 2) & 3;
  int x = k & 3;
  float v = h[((b * 4 + y) * 4 + x) * 512 + c];
  v = v * ab[c] + ab[512 + c];
  out[idx] = (v >= 0.f) ? (int8_t)1 : (int8_t)-1;
}

__global__ void maxpool_kernel(const float* __restrict__ in, int B, int H, int W, int C,
                               float* __restrict__ out) {
  int Ho = H >> 1, Wo = W >> 1;
  long long total = (long long)B * Ho * Wo * C;
  long long idx = (long long)blockIdx.x * blockDim.x + threadIdx.x;
  if (idx >= total) return;
  int c = (int)(idx % C);
  long long t = idx / C;
  int xo = (int)(t % Wo); t /= Wo;
  int yo = (int)(t % Ho); t /= Ho;
  int b = (int)t;
  const float* p = in + (((long long)(b * H + yo * 2) * W + xo * 2) * C + c);
  long long rs = (long long)W * C;
  float m = fmaxf(fmaxf(p[0], p[C]), fmaxf(p[rs], p[rs + C]));
  out[idx] = m;
}

__global__ void wpack_conv_kernel(const float* __restrict__ w, int COUT, int CIN,
                                  int* __restrict__ wpk) {
  int idx = blockIdx.x * blockDim.x + threadIdx.x;
  int kcN = CIN >> 6;
  int nLanes = (COUT >> 4) * 9 * kcN * 32;
  if (idx >= nLanes) return;
  int lane = idx & 31;
  int t = idx >> 5;
  int kc = t % kcN; t /= kcN;
  int tap = t % 9;  t /= 9;
  int nt = t;
  int hi = lane >> 4;
  int n = nt * 16 + (lane & 15);
  int ky = tap / 3, kx = tap % 3;
  int* dst = wpk + (size_t)idx * 8;
  for (int i = 0; i < 8; ++i) {
    int kb = (kc << 6) + kbaseB(i, hi);
    unsigned v = 0;
    for (int j = 0; j < 4; ++j) {
      int cin = kb + j;
      float wv = w[((n * CIN + cin) * 3 + ky) * 3 + kx];
      unsigned byte = (wv >= 0.f) ? 0x01u : 0xFFu;
      v |= byte << (8 * j);
    }
    dst[i] = (int)v;
  }
}

__global__ void wpack_fc_kernel(const float* __restrict__ w, int N, int K,
                                int* __restrict__ wpk) {
  int idx = blockIdx.x * blockDim.x + threadIdx.x;
  int kcN = K >> 6;
  int nLanes = (N >> 4) * kcN * 32;
  if (idx >= nLanes) return;
  int lane = idx & 31;
  int t = idx >> 5;
  int kc = t % kcN; t /= kcN;
  int nt = t;
  int hi = lane >> 4;
  int n = nt * 16 + (lane & 15);
  int* dst = wpk + (size_t)idx * 8;
  for (int i = 0; i < 8; ++i) {
    int kb = (kc << 6) + kbaseB(i, hi);
    unsigned v = 0;
    for (int j = 0; j < 4; ++j) {
      float wv = w[(size_t)n * K + kb + j];
      unsigned byte = (wv >= 0.f) ? 0x01u : 0xFFu;
      v |= byte << (8 * j);
    }
    dst[i] = (int)v;
  }
}

__global__ void bconv_kernel(const int8_t* __restrict__ apk, const int* __restrict__ wpk,
                             float* __restrict__ out, int B, int H, int W, int CIN, int COUT) {
  const int lane = threadIdx.x & 31;
  const int wave = blockIdx.x * (blockDim.x >> 5) + (threadIdx.x >> 5);
  const int nNT = COUT >> 6;
  const int nMT = (B * H * W) >> 4;
  if (wave >= nNT * nMT) return;
  const int mt = wave / nNT;
  const int nt = wave % nNT;
  const int m = lane & 15;
  const int hi = lane >> 4;
  const int p = mt * 16 + m;
  const int xw = p % W;
  const int yw = (p / W) % H;
  const int bb = p / (W * H);
  const int kcN = CIN >> 6;
  const size_t subStride = (size_t)9 * kcN * 256;
  v8i acc0 = {}, acc1 = {}, acc2 = {}, acc3 = {};
  for (int tap = 0; tap < 9; ++tap) {
    const int ys = yw + tap / 3 - 1;
    const int xs = xw + tap % 3 - 1;
    const bool valid = (ys >= 0) && (ys < H) && (xs >= 0) && (xs < W);
    const int ysc = min(max(ys, 0), H - 1);
    const int xsc = min(max(xs, 0), W - 1);
    const int8_t* arow = apk + (size_t)((bb * H + ysc) * W + xsc) * CIN + (hi << 3);
    const int* wb = wpk + ((((size_t)nt * 4) * 9 + tap) * kcN) * 256 + lane * 8;
    for (int kc = 0; kc < kcN; ++kc) {
      v8i a;
#pragma unroll
      for (int g = 0; g < 4; ++g) {
        unsigned long long v =
            *(const unsigned long long*)(arow + (kc << 6) + (g << 4));
        v = valid ? v : 0ull;
        a[2 * g]     = (int)(unsigned)v;
        a[2 * g + 1] = (int)(unsigned)(v >> 32);
      }
      const int* b0 = wb + (size_t)kc * 256;
      v8i bv0 = *(const v8i*)(b0);
      v8i bv1 = *(const v8i*)(b0 + subStride);
      v8i bv2 = *(const v8i*)(b0 + 2 * subStride);
      v8i bv3 = *(const v8i*)(b0 + 3 * subStride);
      acc0 = __builtin_amdgcn_wmma_i32_16x16x64_iu8(true, a, true, bv0, acc0, false, false);
      acc1 = __builtin_amdgcn_wmma_i32_16x16x64_iu8(true, a, true, bv1, acc1, false, false);
      acc2 = __builtin_amdgcn_wmma_i32_16x16x64_iu8(true, a, true, bv2, acc2, false, false);
      acc3 = __builtin_amdgcn_wmma_i32_16x16x64_iu8(true, a, true, bv3, acc3, false, false);
    }
  }
  const int ncol = lane & 15;
#pragma unroll
  for (int r = 0; r < 8; ++r) {
    float* o = out + (size_t)(mt * 16 + r + 8 * hi) * COUT + (nt << 6) + ncol;
    o[0]  = (float)acc0[r];
    o[16] = (float)acc1[r];
    o[32] = (float)acc2[r];
    o[48] = (float)acc3[r];
  }
}

__global__ void bfc_kernel(const int8_t* __restrict__ apk, const int* __restrict__ wpk,
                           float* __restrict__ out, int M, int N, int K) {
  const int lane = threadIdx.x & 31;
  const int wave = blockIdx.x * (blockDim.x >> 5) + (threadIdx.x >> 5);
  const int nNT = N >> 6;
  const int nMT = M >> 4;
  if (wave >= nNT * nMT) return;
  const int mt = wave / nNT;
  const int nt = wave % nNT;
  const int m = lane & 15;
  const int hi = lane >> 4;
  const int kcN = K >> 6;
  const size_t subStride = (size_t)kcN * 256;
  const int8_t* arow = apk + (size_t)(mt * 16 + m) * K + (hi << 3);
  const int* wb = wpk + ((size_t)nt * 4) * subStride + lane * 8;
  v8i acc0 = {}, acc1 = {}, acc2 = {}, acc3 = {};
  for (int kc = 0; kc < kcN; ++kc) {
    v8i a;
#pragma unroll
    for (int g = 0; g < 4; ++g) {
      unsigned long long v =
          *(const unsigned long long*)(arow + (kc << 6) + (g << 4));
      a[2 * g]     = (int)(unsigned)v;
      a[2 * g + 1] = (int)(unsigned)(v >> 32);
    }
    const int* b0 = wb + (size_t)kc * 256;
    v8i bv0 = *(const v8i*)(b0);
    v8i bv1 = *(const v8i*)(b0 + subStride);
    v8i bv2 = *(const v8i*)(b0 + 2 * subStride);
    v8i bv3 = *(const v8i*)(b0 + 3 * subStride);
    acc0 = __builtin_amdgcn_wmma_i32_16x16x64_iu8(true, a, true, bv0, acc0, false, false);
    acc1 = __builtin_amdgcn_wmma_i32_16x16x64_iu8(true, a, true, bv1, acc1, false, false);
    acc2 = __builtin_amdgcn_wmma_i32_16x16x64_iu8(true, a, true, bv2, acc2, false, false);
    acc3 = __builtin_amdgcn_wmma_i32_16x16x64_iu8(true, a, true, bv3, acc3, false, false);
  }
  const int ncol = lane & 15;
#pragma unroll
  for (int r = 0; r < 8; ++r) {
    float* o = out + (size_t)(mt * 16 + r + 8 * hi) * N + (nt << 6) + ncol;
    o[0]  = (float)acc0[r];
    o[16] = (float)acc1[r];
    o[32] = (float)acc2[r];
    o[48] = (float)acc3[r];
  }
}

__global__ void bnrelu_kernel(const float* __restrict__ in, const float* __restrict__ ab,
                              int n, int C, float* __restrict__ out) {
  int idx = blockIdx.x * blockDim.x + threadIdx.x;
  if (idx >= n) return;
  int c = idx % C;
  float v = in[idx] * ab[c] + ab[C + c];
  out[idx] = fmaxf(v, 0.f);
}

__global__ void lastlin_kernel(const float* __restrict__ h, const float* __restrict__ lw,
                               const float* __restrict__ lb, float* __restrict__ out) {
  int idx = blockIdx.x * blockDim.x + threadIdx.x;
  if (idx >= 2560) return;
  int j = idx % 10, i = idx / 10;
  float acc = lb[j];
  for (int k = 0; k < 1024; ++k) acc += h[i * 1024 + k] * lw[j * 1024 + k];
  out[idx] = acc;
}

__global__ void bnlast_kernel(const float* __restrict__ lin, const float* __restrict__ g,
                              const float* __restrict__ b, float* __restrict__ out) {
  __shared__ float s[10], sq[10];
  if (threadIdx.x < 10) { s[threadIdx.x] = 0.f; sq[threadIdx.x] = 0.f; }
  __syncthreads();
  float v[10];
  for (int j = 0; j < 10; ++j) {
    v[j] = lin[threadIdx.x * 10 + j];
    atomicAdd(&s[j], v[j]);
    atomicAdd(&sq[j], v[j] * v[j]);
  }
  __syncthreads();
  for (int j = 0; j < 10; ++j) {
    float m = s[j] * (1.f / 256.f);
    float var = sq[j] * (1.f / 256.f) - m * m;
    out[threadIdx.x * 10 + j] = (v[j] - m) * rsqrtf(var + EPSV) * g[j] + b[j];
  }
}

extern "C" void kernel_launch(void* const* d_in, const int* in_sizes, int n_in,
                              void* d_out, int out_size, void* d_ws, size_t ws_size,
                              hipStream_t stream) {
  (void)in_sizes; (void)n_in; (void)out_size; (void)ws_size;
  const float* x    = (const float*)d_in[0];
  const float* c1w  = (const float*)d_in[1];
  const float* c1b  = (const float*)d_in[2];
  const float* bw1  = (const float*)d_in[3];
  const float* g1a  = (const float*)d_in[4];
  const float* b1a  = (const float*)d_in[5];
  const float* g1b  = (const float*)d_in[6];
  const float* b1b  = (const float*)d_in[7];
  const float* bw2a = (const float*)d_in[8];
  const float* g2a  = (const float*)d_in[9];
  const float* b2a  = (const float*)d_in[10];
  const float* bw2b = (const float*)d_in[11];
  const float* g2b  = (const float*)d_in[12];
  const float* b2b  = (const float*)d_in[13];
  const float* bw3a = (const float*)d_in[14];
  const float* g3a  = (const float*)d_in[15];
  const float* b3a  = (const float*)d_in[16];
  const float* bw3b = (const float*)d_in[17];
  const float* g3b  = (const float*)d_in[18];
  const float* b3b  = (const float*)d_in[19];
  const float* fw1  = (const float*)d_in[20];
  const float* g4a  = (const float*)d_in[21];
  const float* b4a  = (const float*)d_in[22];
  const float* fw2  = (const float*)d_in[23];
  const float* g4b  = (const float*)d_in[24];
  const float* b4b  = (const float*)d_in[25];
  const float* lw   = (const float*)d_in[26];
  const float* lb   = (const float*)d_in[27];
  const float* g5   = (const float*)d_in[28];
  const float* b5   = (const float*)d_in[29];

  char* ws = (char*)d_ws;
  float*  BUF0 = (float*)(ws + 0ull);
  float*  BUF1 = (float*)(ws + 134217728ull);
  int8_t* PK   = (int8_t*)(ws + 268435456ull);
  int*    WB1  = (int*)(ws + 301989888ull);
  int*    WB2a = (int*)(ws + 302137344ull);
  int*    WB2b = (int*)(ws + 302432256ull);
  int*    WB3a = (int*)(ws + 303022080ull);
  int*    WB3b = (int*)(ws + 304201728ull);
  int*    WF1  = (int*)(ws + 306561024ull);
  int*    WF2  = (int*)(ws + 314949632ull);
  float*  STAT = (float*)(ws + 315998208ull);
  float*  AB   = (float*)(ws + 316006400ull);
  float*  LIN  = (float*)(ws + 316014592ull);

  const int Bt = 256;

  auto bn_stats = [&](const float* h, long long n, int C, const float* g, const float* bet) {
    hipMemsetAsync(STAT, 0, 2 * C * sizeof(float), stream);
    stats_kernel<<<2048, 256, 0, stream>>>(h, n, C, STAT);
    bnfin_kernel<<<(C + 63) / 64, 64, 0, stream>>>(STAT, g, bet,
                                                   (float)((double)C / (double)n), C, AB);
  };
  auto bn_pack = [&](const float* h, long long n, int C, const float* g, const float* bet,
                     int8_t* dst) {
    bn_stats(h, n, C, g, bet);
    pack_kernel<<<(int)((n + 255) / 256), 256, 0, stream>>>(h, AB, n, C, dst);
  };

  wpack_conv_kernel<<<(128 * 128 * 9 / 32 + 255) / 256, 256, 0, stream>>>(bw1, 128, 128, WB1);
  wpack_conv_kernel<<<(256 * 128 * 9 / 32 + 255) / 256, 256, 0, stream>>>(bw2a, 256, 128, WB2a);
  wpack_conv_kernel<<<(256 * 256 * 9 / 32 + 255) / 256, 256, 0, stream>>>(bw2b, 256, 256, WB2b);
  wpack_conv_kernel<<<(512 * 256 * 9 / 32 + 255) / 256, 256, 0, stream>>>(bw3a, 512, 256, WB3a);
  wpack_conv_kernel<<<(512 * 512 * 9 / 32 + 255) / 256, 256, 0, stream>>>(bw3b, 512, 512, WB3b);
  wpack_fc_kernel<<<(1024 * 8192 / 32 + 255) / 256, 256, 0, stream>>>(fw1, 1024, 8192, WF1);
  wpack_fc_kernel<<<(1024 * 1024 / 32 + 255) / 256, 256, 0, stream>>>(fw2, 1024, 1024, WF2);

  conv1_kernel<<<(Bt * 32 * 32 * 128) / 256, 256, 0, stream>>>(x, c1w, c1b, BUF0, Bt);
  bn_pack(BUF0, (long long)Bt * 32 * 32 * 128, 128, g1a, b1a, PK);
  { int waves = ((Bt * 32 * 32) / 16) * (128 / 64);
    bconv_kernel<<<waves / 8, 256, 0, stream>>>(PK, WB1, BUF1, Bt, 32, 32, 128, 128); }
  maxpool_kernel<<<(int)(((long long)Bt * 16 * 16 * 128 + 255) / 256), 256, 0, stream>>>(
      BUF1, Bt, 32, 32, 128, BUF0);
  bn_pack(BUF0, (long long)Bt * 16 * 16 * 128, 128, g1b, b1b, PK);

  { int waves = ((Bt * 16 * 16) / 16) * (256 / 64);
    bconv_kernel<<<waves / 8, 256, 0, stream>>>(PK, WB2a, BUF1, Bt, 16, 16, 128, 256); }
  bn_pack(BUF1, (long long)Bt * 16 * 16 * 256, 256, g2a, b2a, PK);
  { int waves = ((Bt * 16 * 16) / 16) * (256 / 64);
    bconv_kernel<<<waves / 8, 256, 0, stream>>>(PK, WB2b, BUF0, Bt, 16, 16, 256, 256); }
  maxpool_kernel<<<(int)(((long long)Bt * 8 * 8 * 256 + 255) / 256), 256, 0, stream>>>(
      BUF0, Bt, 16, 16, 256, BUF1);
  bn_pack(BUF1, (long long)Bt * 8 * 8 * 256, 256, g2b, b2b, PK);

  { int waves = ((Bt * 8 * 8) / 16) * (512 / 64);
    bconv_kernel<<<waves / 8, 256, 0, stream>>>(PK, WB3a, BUF0, Bt, 8, 8, 256, 512); }
  bn_pack(BUF0, (long long)Bt * 8 * 8 * 512, 512, g3a, b3a, PK);
  { int waves = ((Bt * 8 * 8) / 16) * (512 / 64);
    bconv_kernel<<<waves / 8, 256, 0, stream>>>(PK, WB3b, BUF1, Bt, 8, 8, 512, 512); }
  maxpool_kernel<<<(int)(((long long)Bt * 4 * 4 * 512 + 255) / 256), 256, 0, stream>>>(
      BUF1, Bt, 8, 8, 512, BUF0);

  bn_stats(BUF0, (long long)Bt * 4 * 4 * 512, 512, g3b, b3b);
  pack_fc1_kernel<<<(Bt * 8192) / 256, 256, 0, stream>>>(BUF0, AB, Bt, PK);
  { int waves = (Bt / 16) * (1024 / 64);
    bfc_kernel<<<waves / 8, 256, 0, stream>>>(PK, WF1, BUF1, Bt, 1024, 8192); }
  bn_pack(BUF1, (long long)Bt * 1024, 1024, g4a, b4a, PK);
  { int waves = (Bt / 16) * (1024 / 64);
    bfc_kernel<<<waves / 8, 256, 0, stream>>>(PK, WF2, BUF0, Bt, 1024, 1024); }
  bn_stats(BUF0, (long long)Bt * 1024, 1024, g4b, b4b);
  bnrelu_kernel<<<(Bt * 1024) / 256, 256, 0, stream>>>(BUF0, AB, Bt * 1024, 1024, BUF1);

  lastlin_kernel<<<(2560 + 255) / 256, 256, 0, stream>>>(BUF1, lw, lb, LIN);
  bnlast_kernel<<<1, 256, 0, stream>>>(LIN, g5, b5, (float*)d_out);
}
